// RANet_26044681683637
// MI455X (gfx1250) — matrix operand order repaired and checked by reading, not compiled, not run on hardware
//
#include <hip/hip_runtime.h>
#include <hip/hip_bf16.h>

// ---------------- problem constants (fixed by the reference harness) --------
#define F_IN   9
#define DD     256      // D
#define DD2    512      // D2
#define NPG    512      // nodes per graph
#define PAIRS  16
#define BLK    1024     // 2*NPG  (pair block)

typedef __attribute__((ext_vector_type(16))) _Float16 v16h;
typedef __attribute__((ext_vector_type(8)))  _Float16 v8h;
typedef __attribute__((ext_vector_type(8)))  float    v8f;

enum { ACT_NONE = 0, ACT_RELU = 1, ACT_TANH = 2 };

// ---------------------------------------------------------------------------
// Register-blocked batched GEMM:  C[b] = act( scale * A[b] @ Bt[b]^T + bias )
//   A  : row-major  M x K   (f16)
//   Bt : row-major  N x K   (f16)  == B^T, so C[m][n] = sum_k A[m][k]*Bt[n][k]
//   C  : row-major  M x N   (f32)
// One wave32 computes a (16*BM) x (16*BN) macro-tile: BM*BN WMMA accumulators,
// BM+BN fragment loads per K-step -> 8 v_wmma per 6 fragment loads at <4,2>
// (24 B of operand traffic per 16 KFLOP wmma, vs 64 B unblocked).
// Fragment layouts follow CDNA5 ISA 7.12.2:
//   A frag lane(r,h): row r, halves V0-3 = K[h*8 .. h*8+8), V4-7 = K[16+h*8 ..)
//   B frag lane(r,h): col r, V0-3 = K[h*8 .. h*8+8), V4-7 = K[16+h*8 ..)
// ---------------------------------------------------------------------------
template<int BM, int BN>
__global__ __launch_bounds__(256)
void k_wmma_gemm(const _Float16* __restrict__ A, const _Float16* __restrict__ Bt,
                 const float* __restrict__ bias, float* __restrict__ C,
                 int M, int K, int Nout,
                 long aBatch, long bBatch, long cBatch,
                 float scale, int act)
{
    const int batch  = blockIdx.y;
    const int wave   = blockIdx.x * 8 + (threadIdx.x >> 5);
    const int tilesN = Nout / (16 * BN);
    const int tilesM = M / (16 * BM);
    if (wave >= tilesM * tilesN) return;           // whole wave exits: EXEC stays all-1s
    const int tm = wave / tilesN;
    const int tn = wave % tilesN;
    const int lane = threadIdx.x & 31;
    const int half = lane >> 4;
    const int r    = lane & 15;

    const _Float16* ap = A  + (size_t)batch * aBatch + (size_t)(tm * BM * 16 + r) * K;
    const _Float16* bp = Bt + (size_t)batch * bBatch + (size_t)(tn * BN * 16 + r) * K;

    v8f acc[BM][BN] = {};
    for (int k0 = 0; k0 < K; k0 += 32) {
        v16h a[BM], b[BN];
        #pragma unroll
        for (int i = 0; i < BM; ++i) {
            const _Float16* p = ap + (size_t)i * 16 * K + k0;
            v8h lo = *(const v8h*)(p + half * 8);
            v8h hi = *(const v8h*)(p + 16 + half * 8);
            a[i] = __builtin_shufflevector(lo, hi, 0,1,2,3,4,5,6,7,8,9,10,11,12,13,14,15);
        }
        #pragma unroll
        for (int j = 0; j < BN; ++j) {
            const _Float16* p = bp + (size_t)j * 16 * K + k0;
            v8h lo = *(const v8h*)(p + half * 8);
            v8h hi = *(const v8h*)(p + 16 + half * 8);
            b[j] = __builtin_shufflevector(lo, hi, 0,1,2,3,4,5,6,7,8,9,10,11,12,13,14,15);
        }
        #pragma unroll
        for (int i = 0; i < BM; ++i)
            #pragma unroll
            for (int j = 0; j < BN; ++j)
                acc[i][j] = __builtin_amdgcn_wmma_f32_16x16x32_f16(
                    false, a[i], false, b[j], (short)0, acc[i][j], false, false);
    }

    float* cp = C + (size_t)batch * cBatch;
    #pragma unroll
    for (int j = 0; j < BN; ++j) {
        const int n = (tn * BN + j) * 16 + r;
        const float bv = bias ? bias[n] : 0.0f;
        #pragma unroll
        for (int i = 0; i < BM; ++i) {
            const int mbase = (tm * BM + i) * 16 + half * 8;   // ISA C/D layout: VGPR v -> M = v + 8*half
            #pragma unroll
            for (int v = 0; v < 8; ++v) {
                float x = acc[i][j][v] * scale + bv;
                if (act == ACT_RELU)      x = fmaxf(x, 0.0f);
                else if (act == ACT_TANH) x = tanhf(x);
                cp[(size_t)(mbase + v) * Nout + n] = x;
            }
        }
    }
}

// ------------------------- small helper kernels ----------------------------
__global__ void k_f32_to_f16(const float* __restrict__ in, _Float16* __restrict__ out, long n)
{
    long i = (long)blockIdx.x * blockDim.x + threadIdx.x;
    if (i < n) out[i] = (_Float16)in[i];
}

// W is K x N row-major f32; write Wt = N x K row-major f16
__global__ void k_transpose_f16(const float* __restrict__ W, _Float16* __restrict__ Wt,
                                int K, int N)
{
    long i = (long)blockIdx.x * blockDim.x + threadIdx.x;
    if (i >= (long)K * N) return;
    int k = (int)(i / N), n = (int)(i % N);
    Wt[(size_t)n * K + k] = (_Float16)W[i];
}

// v: [N_nodes][DD] f32 (pair-contiguous)  ->  vt: [PAIRS][DD][BLK] f16
__global__ void k_vtranspose(const float* __restrict__ v, _Float16* __restrict__ vt)
{
    long i = (long)blockIdx.x * blockDim.x + threadIdx.x;
    if (i >= (long)PAIRS * DD * BLK) return;
    int p = (int)(i / ((long)DD * BLK));
    int rem = (int)(i % ((long)DD * BLK));
    int d = rem / BLK, j = rem % BLK;
    vt[i] = (_Float16)v[((size_t)p * BLK + j) * DD + d];
}

// XW = x @ w_in   (K = 9, done scalar)
__global__ void k_in_proj(const float* __restrict__ x, const float* __restrict__ w,
                          float* __restrict__ xw, int n_nodes)
{
    long i = (long)blockIdx.x * blockDim.x + threadIdx.x;
    if (i >= (long)n_nodes * DD) return;
    int node = (int)(i / DD), d = (int)(i % DD);
    float s = 0.0f;
    #pragma unroll
    for (int j = 0; j < F_IN; ++j) s += x[(size_t)node * F_IN + j] * w[j * DD + d];
    xw[i] = s;
}

__global__ void k_set1(float* __restrict__ deg, int n)
{
    int i = blockIdx.x * blockDim.x + threadIdx.x;
    if (i < n) deg[i] = 1.0f;                      // self loop
}
__global__ void k_deg_count(const int* __restrict__ dst, float* __restrict__ deg, long E)
{
    long i = (long)blockIdx.x * blockDim.x + threadIdx.x;
    if (i < E) atomicAdd(&deg[dst[i]], 1.0f);
}
__global__ void k_rsqrt(float* __restrict__ deg, int n)
{
    int i = blockIdx.x * blockDim.x + threadIdx.x;
    if (i < n) deg[i] = rsqrtf(deg[i]);            // deg -> dis
}

// agg[dst] += dis[src]*dis[dst] * XW[src]   (one thread = one edge x 4 features)
__global__ void k_edge_agg(const int* __restrict__ src, const int* __restrict__ dst,
                           const float* __restrict__ dis, const float* __restrict__ XW,
                           float* __restrict__ agg, long E)
{
    long i = (long)blockIdx.x * blockDim.x + threadIdx.x;
    if (i >= E * (DD / 4)) return;
    long e = i >> 6;                               // DD/4 == 64
    int  c = (int)(i & 63);
    int  s = src[e], t = dst[e];
    float nrm = dis[s] * dis[t];
    float4 v = ((const float4*)XW)[(size_t)s * (DD / 4) + c];
    float* a = agg + (size_t)t * DD + c * 4;
    atomicAdd(a + 0, nrm * v.x);
    atomicAdd(a + 1, nrm * v.y);
    atomicAdd(a + 2, nrm * v.z);
    atomicAdd(a + 3, nrm * v.w);
}

// h = tanh(agg + dis^2 * XW + b)   (adds self-loop term + bias)
__global__ void k_gcn_finish(const float* __restrict__ agg, const float* __restrict__ XW,
                             const float* __restrict__ dis, const float* __restrict__ b,
                             float* __restrict__ h, int n_nodes)
{
    long i = (long)blockIdx.x * blockDim.x + threadIdx.x;
    if (i >= (long)n_nodes * DD) return;
    int node = (int)(i / DD), d = (int)(i % DD);
    float self = dis[node] * dis[node];
    h[i] = tanhf(agg[i] + self * XW[i] + b[d]);
}

// masked softmax per (pair,row); writes f16 probabilities (masked -> 0)
__global__ __launch_bounds__(256)
void k_softmax(const float* __restrict__ scores, _Float16* __restrict__ probs)
{
    const int row = blockIdx.x;
    const int p = row >> 10, i = row & (BLK - 1);
    const int ih = i >> 9;                         // i / NPG
    const float* srow = scores + ((size_t)p << 20) + ((size_t)i << 10);
    _Float16*    prow = probs  + ((size_t)p << 20) + ((size_t)i << 10);
    __shared__ float red[8];
    const int tid = threadIdx.x, lane = tid & 31, wv = tid >> 5;

    float mx = -3.4e38f;
    for (int j = tid; j < BLK; j += 256)
        if ((j >> 9) != ih) mx = fmaxf(mx, srow[j]);
    for (int o = 16; o > 0; o >>= 1) mx = fmaxf(mx, __shfl_down(mx, o, 32));
    if (lane == 0) red[wv] = mx;
    __syncthreads();
    if (tid == 0) { float m = red[0]; for (int q = 1; q < 8; ++q) m = fmaxf(m, red[q]); red[0] = m; }
    __syncthreads();
    mx = red[0];
    __syncthreads();

    float sum = 0.0f;
    for (int j = tid; j < BLK; j += 256)
        if ((j >> 9) != ih) sum += __expf(srow[j] - mx);
    for (int o = 16; o > 0; o >>= 1) sum += __shfl_down(sum, o, 32);
    if (lane == 0) red[wv] = sum;
    __syncthreads();
    if (tid == 0) { float s = 0; for (int q = 0; q < 8; ++q) s += red[q]; red[0] = s; }
    __syncthreads();
    const float inv = 1.0f / red[0];

    for (int j = tid; j < BLK; j += 256)
        prow[j] = ((j >> 9) != ih) ? (_Float16)(__expf(srow[j] - mx) * inv) : (_Float16)0.0f;
}

// fc3 (h2 is [N][32]) dot + per-graph mean + sigmoid
__global__ __launch_bounds__(256)
void k_fc3_pool(const float* __restrict__ h2, const float* __restrict__ w,
                const float* __restrict__ b, float* __restrict__ out)
{
    const int g = blockIdx.x;                      // 32 graphs
    const int tid = threadIdx.x, lane = tid & 31, wv = tid >> 5;
    __shared__ float red[8];
    float acc = 0.0f;
    for (int t = tid; t < NPG; t += 256) {
        const float* row = h2 + ((size_t)g * NPG + t) * 32;
        float s = b[0];
        #pragma unroll
        for (int j = 0; j < 32; ++j) s += row[j] * w[j];
        acc += s;
    }
    for (int o = 16; o > 0; o >>= 1) acc += __shfl_down(acc, o, 32);
    if (lane == 0) red[wv] = acc;
    __syncthreads();
    if (tid == 0) {
        float s = 0; for (int q = 0; q < 8; ++q) s += red[q];
        float mean = s / (float)NPG;
        out[g] = 1.0f / (1.0f + __expf(-mean));
    }
}

// ---------------------------------------------------------------------------
extern "C" void kernel_launch(void* const* d_in, const int* in_sizes, int n_in,
                              void* d_out, int out_size, void* d_ws, size_t ws_size,
                              hipStream_t stream)
{
    const float* x        = (const float*)d_in[0];
    const int*   ei       = (const int*)  d_in[1];   // [2][E]
    const float* w_in     = (const float*)d_in[3];
    const float* b_in     = (const float*)d_in[4];
    const float* conv_ws_ = (const float*)d_in[5];   // [3][D][D]
    const float* conv_bs_ = (const float*)d_in[6];   // [3][D]
    const float* mlp_w1   = (const float*)d_in[7];   // [3][D][D2]
    const float* mlp_b1   = (const float*)d_in[8];
    const float* mlp_w2   = (const float*)d_in[9];   // [3][D2][D2]
    const float* mlp_b2   = (const float*)d_in[10];
    const float* mlp_w3   = (const float*)d_in[11];  // [3][D2][D]
    const float* mlp_b3   = (const float*)d_in[12];
    const float* fc1_w    = (const float*)d_in[13];
    const float* fc1_b    = (const float*)d_in[14];
    const float* fc2_w    = (const float*)d_in[15];
    const float* fc2_b    = (const float*)d_in[16];
    const float* fc3_w    = (const float*)d_in[17];
    const float* fc3_b    = (const float*)d_in[18];
    float* out = (float*)d_out;

    const int  N = in_sizes[0] / F_IN;               // 16384
    const long E = in_sizes[1] / 2;                  // 524288
    const int* src = ei;
    const int* dst = ei + E;

    // ----- workspace layout (bump allocator, 256B aligned) -----
    char* ws = (char*)d_ws;
    size_t off = 0;
    auto alloc = [&](size_t bytes) -> void* {
        void* p = ws + off;
        off = (off + bytes + 255) & ~(size_t)255;
        return p;
    };
    float*     deg  = (float*)    alloc((size_t)N * 4);                 // then holds dis
    float*     XW   = (float*)    alloc((size_t)N * DD * 4);
    float*     h    = (float*)    alloc((size_t)N * DD * 4);
    _Float16*  h16  = (_Float16*) alloc((size_t)N * DD * 2);
    _Float16*  t16  = (_Float16*) alloc((size_t)N * DD2 * 2);
    _Float16*  w16  = (_Float16*) alloc((size_t)DD2 * DD2 * 2);
    float*     t1   = (float*)    alloc((size_t)N * DD2 * 4);
    float*     t2   = (float*)    alloc((size_t)N * DD2 * 4);           // contiguous after t1
    float*     qkv  = (float*)    alloc((size_t)3 * N * DD * 4);
    _Float16*  k16  = (_Float16*) alloc((size_t)N * DD * 2);
    _Float16*  vt16 = (_Float16*) alloc((size_t)N * DD * 2);
    // aliases over dead regions:
    float*     scores  = t1;                 // 64MB = t1 + t2 (free after MLPs)
    _Float16*  probs16 = (_Float16*)qkv;     // 32MB over dead q/k f32
    _Float16*  q16     = h16;                // h16 free after MLP chain
    float*     agg     = t1;                 // GCN conv scratch
    float*     h1      = t1;                 // fc1 out
    _Float16*  h1_16   = t16;
    float*     h2      = t2;                 // fc2 out [N][32]
    (void)ws_size; (void)n_in; (void)out_size;

    auto blocks = [](long n, int bs) { return (unsigned)((n + bs - 1) / bs); };
    // all GEMMs in this net have M % 64 == 0 and Nout % 32 == 0 -> one <4,2> config
    auto gemm = [&](const _Float16* A, const _Float16* Bt, const float* bias, float* C,
                    int M, int K, int Nout, int batch,
                    long aB, long bB, long cB, float scale, int act) {
        int waves = (M / 64) * (Nout / 32);
        dim3 grid((waves + 7) / 8, batch);
        k_wmma_gemm<4, 2><<<grid, 256, 0, stream>>>(A, Bt, bias, C, M, K, Nout,
                                                    aB, bB, cB, scale, act);
    };

    // ----- degrees -> dis = rsqrt(1 + indeg) -----
    k_set1     <<<blocks(N, 256), 256, 0, stream>>>(deg, N);
    k_deg_count<<<blocks(E, 256), 256, 0, stream>>>(dst, deg, E);
    k_rsqrt    <<<blocks(N, 256), 256, 0, stream>>>(deg, N);

    // ----- input GCN: h = tanh(agg(x @ w_in) + b_in) -----
    k_in_proj<<<blocks((long)N * DD, 256), 256, 0, stream>>>(x, w_in, XW, N);
    hipMemsetAsync(agg, 0, (size_t)N * DD * 4, stream);
    k_edge_agg<<<blocks(E * (DD / 4), 256), 256, 0, stream>>>(src, dst, deg, XW, agg, E);
    k_gcn_finish<<<blocks((long)N * DD, 256), 256, 0, stream>>>(agg, XW, deg, b_in, h, N);

    // ----- q/k/v MLPs (WMMA GEMMs, ReLU) -----
    k_f32_to_f16<<<blocks((long)N * DD, 256), 256, 0, stream>>>(h, h16, (long)N * DD);
    for (int t = 0; t < 3; ++t) {
        k_transpose_f16<<<blocks((long)DD * DD2, 256), 256, 0, stream>>>(
            mlp_w1 + (size_t)t * DD * DD2, w16, DD, DD2);
        gemm(h16, w16, mlp_b1 + t * DD2, t1, N, DD, DD2, 1, 0, 0, 0, 1.0f, ACT_RELU);

        k_f32_to_f16<<<blocks((long)N * DD2, 256), 256, 0, stream>>>(t1, t16, (long)N * DD2);
        k_transpose_f16<<<blocks((long)DD2 * DD2, 256), 256, 0, stream>>>(
            mlp_w2 + (size_t)t * DD2 * DD2, w16, DD2, DD2);
        gemm(t16, w16, mlp_b2 + t * DD2, t2, N, DD2, DD2, 1, 0, 0, 0, 1.0f, ACT_RELU);

        k_f32_to_f16<<<blocks((long)N * DD2, 256), 256, 0, stream>>>(t2, t16, (long)N * DD2);
        k_transpose_f16<<<blocks((long)DD2 * DD, 256), 256, 0, stream>>>(
            mlp_w3 + (size_t)t * DD2 * DD, w16, DD2, DD);
        gemm(t16, w16, mlp_b3 + t * DD, qkv + (size_t)t * N * DD,
             N, DD2, DD, 1, 0, 0, 0, 1.0f, ACT_RELU);
    }

    // ----- cross attention over 16 pair-blocks of 1024 nodes -----
    k_f32_to_f16<<<blocks((long)N * DD, 256), 256, 0, stream>>>(qkv, q16, (long)N * DD);
    k_f32_to_f16<<<blocks((long)N * DD, 256), 256, 0, stream>>>(qkv + (size_t)N * DD, k16, (long)N * DD);
    k_vtranspose<<<blocks((long)PAIRS * DD * BLK, 256), 256, 0, stream>>>(qkv + (size_t)2 * N * DD, vt16);

    // scores[p] = (q[p] @ k[p]^T) / 16    (both operands row-major [node][d])
    gemm(q16, k16, nullptr, scores, BLK, DD, BLK, PAIRS,
         (long)BLK * DD, (long)BLK * DD, (long)BLK * BLK, 1.0f / 16.0f, ACT_NONE);
    k_softmax<<<PAIRS * BLK, 256, 0, stream>>>(scores, probs16);
    // h[p] = probs[p] @ v[p]
    gemm(probs16, vt16, nullptr, h, BLK, BLK, DD, PAIRS,
         (long)BLK * BLK, (long)DD * BLK, (long)BLK * DD, 1.0f, ACT_NONE);

    // ----- 3 extra GCN convs: h = tanh(agg(h @ W) + b) -----
    for (int i = 0; i < 3; ++i) {
        k_f32_to_f16<<<blocks((long)N * DD, 256), 256, 0, stream>>>(h, h16, (long)N * DD);
        k_transpose_f16<<<blocks((long)DD * DD, 256), 256, 0, stream>>>(
            conv_ws_ + (size_t)i * DD * DD, w16, DD, DD);
        gemm(h16, w16, nullptr, XW, N, DD, DD, 1, 0, 0, 0, 1.0f, ACT_NONE);
        hipMemsetAsync(agg, 0, (size_t)N * DD * 4, stream);
        k_edge_agg<<<blocks(E * (DD / 4), 256), 256, 0, stream>>>(src, dst, deg, XW, agg, E);
        k_gcn_finish<<<blocks((long)N * DD, 256), 256, 0, stream>>>(
            agg, XW, deg, conv_bs_ + (size_t)i * DD, h, N);
    }

    // ----- head: fc1 (tanh), fc2 (tanh), fc3 + mean-pool + sigmoid -----
    k_f32_to_f16<<<blocks((long)N * DD, 256), 256, 0, stream>>>(h, h16, (long)N * DD);
    k_transpose_f16<<<blocks((long)DD * DD, 256), 256, 0, stream>>>(fc1_w, w16, DD, DD);
    gemm(h16, w16, fc1_b, h1, N, DD, DD, 1, 0, 0, 0, 1.0f, ACT_TANH);

    k_f32_to_f16<<<blocks((long)N * DD, 256), 256, 0, stream>>>(h1, h1_16, (long)N * DD);
    k_transpose_f16<<<blocks((long)DD * 32, 256), 256, 0, stream>>>(fc2_w, w16, DD, 32);
    gemm(h1_16, w16, fc2_b, h2, N, DD, 32, 1, 0, 0, 0, 1.0f, ACT_TANH);

    k_fc3_pool<<<2 * PAIRS, 256, 0, stream>>>(h2, fc3_w, fc3_b, out);
}
